// TopNRouter_3393024163883
// MI455X (gfx1250) — compile-verified, matrix operand order checked by reading, not run on hardware
//
#include <hip/hip_runtime.h>

typedef __bf16 bf16_t;
typedef __attribute__((ext_vector_type(16))) __bf16 v16bf;
typedef __attribute__((ext_vector_type(8)))  __bf16 v8bf;
typedef __attribute__((ext_vector_type(4)))  __bf16 v4bf;
typedef __attribute__((ext_vector_type(8)))  float  v8f;
typedef __attribute__((ext_vector_type(4)))  float  v4f;
typedef __attribute__((ext_vector_type(4)))  int    v4i;

// async global->LDS builtin signature (probe-verified):
//   param0: addrspace(1) v4i*  (global source)
//   param1: addrspace(3) v4i*  (LDS destination)
typedef __attribute__((address_space(1))) v4i g_v4i;
typedef __attribute__((address_space(3))) v4i l_v4i;

#define HIDDEN    4096
#define NEXPERT   64
#define NTOK      16384
#define BLOCK     128                 // 4 waves -> 256 workgroups (device spread)
#define TPW       16                  // tokens per wave (one 16x16 M-tile)
#define TPB       64                  // tokens per block (4 waves)
#define KC        128                 // K-chunk staged in LDS (64x128 bf16 = 16 KB)
#define NCHUNK    (HIDDEN / KC)       // 32
#define STAGE_OPS 8                   // async-load instructions per thread per stage

#if __has_builtin(__builtin_amdgcn_global_load_async_to_lds_b128)
#define ASYNC_LDS 1
#endif

#ifdef ASYNC_LDS
#if __has_builtin(__builtin_amdgcn_s_wait_asynccnt)
#define WAIT_ASYNC(n) __builtin_amdgcn_s_wait_asynccnt(n)
#else
#define WAIT_ASYNC(n) asm volatile("s_wait_asynccnt %0" ::"i"(n) : "memory")
#endif
#else
#define WAIT_ASYNC(n) ((void)0)
#endif

// ---------------------------------------------------------------------------
// Pass 0: convert router weights W [64,4096] f32 -> bf16 (512 KB in d_ws).
// ---------------------------------------------------------------------------
__global__ __launch_bounds__(256) void w_to_bf16(const float* __restrict__ W,
                                                 bf16_t* __restrict__ Wb) {
  int i = blockIdx.x * 256 + threadIdx.x;      // one float4 per thread
  v4f v = ((const v4f*)W)[i];
  v4bf o;
  o[0] = (bf16_t)v[0]; o[1] = (bf16_t)v[1];
  o[2] = (bf16_t)v[2]; o[3] = (bf16_t)v[3];
  ((v4bf*)Wb)[i] = o;
}

// ---------------------------------------------------------------------------
// Stage one K-chunk of W (64 experts x KC k) into LDS, layout [expert][k].
// 128 threads x 8 x 16B async copies = 16 KB, no VGPR round-trip (ASYNCcnt).
// ---------------------------------------------------------------------------
__device__ __forceinline__ void stage_b(const bf16_t* __restrict__ Wb,
                                        bf16_t* __restrict__ lbuf,
                                        int kc, int tid) {
#pragma unroll
  for (int j = 0; j < STAGE_OPS; ++j) {
    int i   = tid * STAGE_OPS + j;     // 1024 x 16B blocks
    int e   = i >> 4;                  // 16 blocks per expert row
    int off = (i & 15) * 8;            // bf16 units
    const bf16_t* src = Wb + (size_t)e * HIDDEN + kc + off;
    bf16_t*       dst = lbuf + e * KC + off;
#ifdef ASYNC_LDS
    __builtin_amdgcn_global_load_async_to_lds_b128((g_v4i*)src, (l_v4i*)dst,
                                                   0, 0);
#else
    *(v8bf*)dst = *(const v8bf*)src;   // fallback through VGPRs
#endif
  }
}

__device__ __forceinline__ v16bf cvt16(v4f a0, v4f a1, v4f a2, v4f a3) {
  v16bf r;
  r[0]  = (bf16_t)a0[0]; r[1]  = (bf16_t)a0[1];
  r[2]  = (bf16_t)a0[2]; r[3]  = (bf16_t)a0[3];
  r[4]  = (bf16_t)a1[0]; r[5]  = (bf16_t)a1[1];
  r[6]  = (bf16_t)a1[2]; r[7]  = (bf16_t)a1[3];
  r[8]  = (bf16_t)a2[0]; r[9]  = (bf16_t)a2[1];
  r[10] = (bf16_t)a2[2]; r[11] = (bf16_t)a2[3];
  r[12] = (bf16_t)a3[0]; r[13] = (bf16_t)a3[1];
  r[14] = (bf16_t)a3[2]; r[15] = (bf16_t)a3[3];
  return r;
}

__device__ __forceinline__ v16bf ldb(const bf16_t* p) {  // two ds_load_b128
  v8bf lo = *(const v8bf*)p;
  v8bf hi = *(const v8bf*)(p + 16);
  return __builtin_shufflevector(lo, hi, 0, 1, 2, 3, 4, 5, 6, 7,
                                 8, 9, 10, 11, 12, 13, 14, 15);
}

// ---------------------------------------------------------------------------
// Pass 1: fused GEMM (M=16384, N=64, K=4096, bf16 WMMA, f32 accum) + top-2.
// Wave tile: 16 tokens x 64 experts. Block: 4 waves = 64 tokens; 256 blocks.
// W chunks double-buffered in LDS via async copies, shared by all 4 waves.
// ---------------------------------------------------------------------------
__global__ __launch_bounds__(BLOCK, 1) void router_gemm_topk(
    const float* __restrict__ X, const bf16_t* __restrict__ Wb,
    float* __restrict__ scores, int* __restrict__ topidx) {
  __shared__ __align__(16) char smem[2 * NEXPERT * KC * 2];  // 32 KB
  bf16_t* bbuf0 = (bf16_t*)smem;
  bf16_t* bbuf1 = (bf16_t*)(smem + NEXPERT * KC * 2);
  float*  lds_logits = (float*)smem;   // aliases B buffers after the K loop

  const int tid  = threadIdx.x;
  const int wave = tid >> 5;
  const int lane = tid & 31;
  const int half = lane >> 4;          // 0: K 0..7/16..23, 1: K 8..15/24..31
  const int l16  = lane & 15;
  const int tok0 = blockIdx.x * TPB + wave * TPW;

  // A: lane holds row m = l16; two 8-float chunks per K-step of 32.
  const float* arow = X + (size_t)(tok0 + l16) * HIDDEN + half * 8;

  v8f c0 = {}, c1 = {}, c2 = {}, c3 = {};

  stage_b(Wb, bbuf0, 0, tid);          // prologue: chunk 0 in flight

  for (int ch = 0; ch < NCHUNK; ++ch) {
    bf16_t* cur = (ch & 1) ? bbuf1 : bbuf0;
    bf16_t* nxt = (ch & 1) ? bbuf0 : bbuf1;
    if (ch + 1 < NCHUNK) {
      stage_b(Wb, nxt, (ch + 1) * KC, tid);  // prefetch next chunk
      WAIT_ASYNC(STAGE_OPS);                 // in-order: current chunk landed
    } else {
      WAIT_ASYNC(0);
    }
    __syncthreads();                         // chunk visible to all waves

    const int kc = ch * KC;
    const bf16_t* rB = cur + l16 * KC + half * 8;   // expert tile stride 16*KC

#pragma unroll
    for (int ks = 0; ks < KC; ks += 32) {
      // A fragment: 4 aligned non-temporal b128 loads + f32->bf16
      v4f a0 = __builtin_nontemporal_load((const v4f*)(arow + kc + ks));
      v4f a1 = __builtin_nontemporal_load((const v4f*)(arow + kc + ks + 4));
      v4f a2 = __builtin_nontemporal_load((const v4f*)(arow + kc + ks + 16));
      v4f a3 = __builtin_nontemporal_load((const v4f*)(arow + kc + ks + 20));
      v16bf av = cvt16(a0, a1, a2, a3);

      // batch all 8 ds_load_b128 first -> one dscnt wait, back-to-back WMMAs
      v16bf b0 = ldb(rB + ks);
      v16bf b1 = ldb(rB + ks + 16 * KC);
      v16bf b2 = ldb(rB + ks + 32 * KC);
      v16bf b3 = ldb(rB + ks + 48 * KC);

      c0 = __builtin_amdgcn_wmma_f32_16x16x32_bf16(false, av, false, b0,
                                                   (short)0, c0, false, false);
      c1 = __builtin_amdgcn_wmma_f32_16x16x32_bf16(false, av, false, b1,
                                                   (short)0, c1, false, false);
      c2 = __builtin_amdgcn_wmma_f32_16x16x32_bf16(false, av, false, b2,
                                                   (short)0, c2, false, false);
      c3 = __builtin_amdgcn_wmma_f32_16x16x32_bf16(false, av, false, b3,
                                                   (short)0, c3, false, false);
    }
    __syncthreads();   // all waves done reading `cur` before it is restaged
  }

  // ---- scatter C (16x16 f32 layout: VGPR r -> M = r + half*8, lane -> N) ----
  float* myl = lds_logits + (wave * TPW + half * 8) * NEXPERT + l16;
#pragma unroll
  for (int r = 0; r < 8; ++r) {
    myl[r * NEXPERT + 0]  = c0[r];
    myl[r * NEXPERT + 16] = c1[r];
    myl[r * NEXPERT + 32] = c2[r];
    myl[r * NEXPERT + 48] = c3[r];
  }
  __syncthreads();

  // ---- top-2 per token (strict '>' == lax.top_k tie-break: lowest index) ----
  if (tid < TPB) {
    const float* row = lds_logits + tid * NEXPERT;
    float best = row[0], second = -INFINITY;
    int bi = 0, si = 0;
#pragma unroll 8
    for (int e = 1; e < NEXPERT; ++e) {
      float v = row[e];
      if (v > best) {
        second = best; si = bi; best = v; bi = e;
      } else if (v > second) {
        second = v; si = e;
      }
    }
    int tok = blockIdx.x * TPB + tid;
    scores[tok * 2 + 0] = best;
    scores[tok * 2 + 1] = second;
    topidx[tok * 2 + 0] = bi;
    topidx[tok * 2 + 1] = si;
  }
}

// ---------------------------------------------------------------------------
extern "C" void kernel_launch(void* const* d_in, const int* in_sizes, int n_in,
                              void* d_out, int out_size, void* d_ws, size_t ws_size,
                              hipStream_t stream) {
  (void)in_sizes; (void)n_in; (void)out_size; (void)ws_size;
  const float* X = (const float*)d_in[0];   // hidden_states [4,4096,4096] f32
  const float* W = (const float*)d_in[1];   // router weights [64,4096] f32

  bf16_t* Wb = (bf16_t*)d_ws;               // 512 KB bf16 weights in scratch
  w_to_bf16<<<(NEXPERT * HIDDEN / 4) / 256, 256, 0, stream>>>(W, Wb);

  float* scores = (float*)d_out;            // [16384,2] f32
  int*   topidx = (int*)d_out + NTOK * 2;   // [16384,2] int32 (tuple order)
  router_gemm_topk<<<NTOK / TPB, BLOCK, 0, stream>>>(X, Wb, scores, topidx);
}
